// Emb_to_Data_3341484556808
// MI455X (gfx1250) — matrix operand order repaired and checked by reading, not compiled, not run on hardware
//
#include <hip/hip_runtime.h>
#include <math.h>

// ---------------------------------------------------------------------------
// Persistent bf16-WMMA LSTM decoder for gfx1250 (MI455X).
//   - One launch, 32 workgroups x 16 batch rows, internal t-loop (no grid sync
//     needed: recurrence is row-local in batch).
//   - Weights packed/padded to bf16 once into workspace (deterministic per call).
//   - All GEMMs: v_wmma_f32_16x16x32_bf16, activations staged in LDS.
// ---------------------------------------------------------------------------

typedef __attribute__((ext_vector_type(16))) __bf16 v16bf;
typedef __attribute__((ext_vector_type(8)))  __bf16 bf16x8;
typedef __attribute__((ext_vector_type(8)))  float  v8f;

#define B_SZ   512
#define T_SZ   200
#define DMD    264     // d_model
#define HL     528     // H_LSTM
#define DI     1024    // d_inner
#define NGATE  2112    // 4*HL
#define K1     1120    // cat1 padded K: trg(288) | pe(288) | h(544)
#define PE0    288
#define H1OFF  576
#define K2     832     // cat2 padded K: h(544) | emb(288)
#define EMB0   544
#define W1K    791     // raw W1 inner dim
#define N2P    272     // W2 rows padded 264 -> 272
#define MROWS  16      // batch rows per workgroup
#define NW     8       // waves per workgroup (256 threads, wave32)

__device__ __forceinline__ float sigm(float x)  { return 1.f / (1.f + __expf(-x)); }
__device__ __forceinline__ float lrelu(float x) { return x > 0.f ? x : 0.01f * x; }

// A-matrix 16x32 bf16 tile: lane&15 = row M; low lanes K {k0..k0+7, k0+16..k0+23},
// high lanes K {k0+8..k0+15, k0+24..k0+31}.  Two 16B loads per lane.
__device__ __forceinline__ v16bf wload_a(const __bf16* base, int ld, int k0, int lane) {
  const int r  = lane & 15;
  const int hi = (lane >> 4) << 3;             // 0 or 8
  const __bf16* p = base + r * ld + k0 + hi;   // 16B aligned (ld % 8 == 0, k0 % 32 == 0)
  union { v16bf v; bf16x8 h[2]; } u;
  u.h[0] = *(const bf16x8*)(p);
  u.h[1] = *(const bf16x8*)(p + 16);
  return u.v;
}

// B-matrix 32x16 bf16 tile from weights stored row-major [N][K] (out[n] = sum_k act[k]*W[n][k]):
// lane&15 = column N; low lanes K {k0..k0+7, k0+16..k0+23}, high lanes K {k0+8..k0+15, k0+24..k0+31}.
__device__ __forceinline__ v16bf wload_b(const __bf16* W, int ld, int n0, int k0, int lane) {
  const int n  = n0 + (lane & 15);
  const int kh = k0 + ((lane >> 4) << 3);      // 0 or 8
  const __bf16* p = W + n * ld + kh;
  union { v16bf v; bf16x8 h[2]; } u;
  u.h[0] = *(const bf16x8*)(p);
  u.h[1] = *(const bf16x8*)(p + 16);
  return u.v;
}

__device__ __forceinline__ v8f wmma_bf16(v16bf a, v16bf b, v8f c) {
  // (neg_a, A, neg_b, B, c_mod, C, reuse_a, reuse_b)
  return __builtin_amdgcn_wmma_f32_16x16x32_bf16(false, a, false, b, (short)0, c, false, false);
}

// ---------------------------------------------------------------------------
// Weight packing (fp32 -> bf16, K-padded layouts matching LDS activation cats)
// ---------------------------------------------------------------------------
__global__ __launch_bounds__(256) void pack_gates_k(const float* __restrict__ Wih,
                                                    const float* __restrict__ Whh,
                                                    __bf16* __restrict__ Wg) {
  int idx = blockIdx.x * 256 + threadIdx.x;
  if (idx >= NGATE * K1) return;
  int n = idx / K1, k = idx - n * K1;
  float v = 0.f;
  if (k < 264)                      v = Wih[n * HL + k];            // trg part
  else if (k >= 288 && k < 552)     v = Wih[n * HL + (k - 24)];     // pe part
  else if (k >= 576 && k < 1104)    v = Whh[n * HL + (k - 576)];    // h part
  Wg[idx] = (__bf16)v;
}

__global__ __launch_bounds__(256) void pack_w1_k(const float* __restrict__ W1,
                                                 __bf16* __restrict__ W1p) {
  int idx = blockIdx.x * 256 + threadIdx.x;
  if (idx >= DI * K2) return;
  int n = idx / K2, k = idx - n * K2;
  float v = 0.f;
  if (k < 528)                      v = W1[n * W1K + k];            // h part
  else if (k >= 544 && k < 807)     v = W1[n * W1K + (k - 16)];     // emb part
  W1p[idx] = (__bf16)v;
}

__global__ __launch_bounds__(256) void pack_wm_k(const float* __restrict__ Wm,
                                                 __bf16* __restrict__ Wmp) {
  int idx = blockIdx.x * 256 + threadIdx.x;
  if (idx >= DI * DI) return;
  Wmp[idx] = (__bf16)Wm[idx];
}

__global__ __launch_bounds__(256) void pack_w2_k(const float* __restrict__ W2,
                                                 __bf16* __restrict__ W2p) {
  int idx = blockIdx.x * 256 + threadIdx.x;
  if (idx >= N2P * DI) return;
  int n = idx / DI, k = idx - n * DI;
  W2p[idx] = (__bf16)((n < DMD) ? W2[n * DI + k] : 0.f);
}

__global__ __launch_bounds__(256) void bias_sum_k(const float* __restrict__ bih,
                                                  const float* __restrict__ bhh,
                                                  float* __restrict__ bg) {
  int idx = blockIdx.x * 256 + threadIdx.x;
  if (idx < NGATE) bg[idx] = bih[idx] + bhh[idx];
}

// ---------------------------------------------------------------------------
// Persistent decoder kernel: 32 blocks x 256 threads; block handles 16 rows.
// ---------------------------------------------------------------------------
__global__ __launch_bounds__(256) void decode_kernel(
    const float*  __restrict__ emb_seq,   // [512][200][264]
    const float*  __restrict__ pos_table, // [205][264]
    const __bf16* __restrict__ Wg,        // [2112][1120]
    const __bf16* __restrict__ W1p,       // [1024][832]
    const __bf16* __restrict__ Wmp,       // [1024][1024]
    const __bf16* __restrict__ W2p,       // [272][1024]
    const float*  __restrict__ biasg,     // [2112]
    const float*  __restrict__ b1,        // [1024]
    const float*  __restrict__ bmv,       // [1024]
    const float*  __restrict__ b2,        // [264]
    float*        __restrict__ out)       // [512][200][264]
{
  __shared__ __attribute__((aligned(16))) __bf16 cat1[MROWS][K1];  // trg|pe|h
  __shared__ __attribute__((aligned(16))) __bf16 cat2[MROWS][K2];  // h|emb
  __shared__ __attribute__((aligned(16))) __bf16 x1s[MROWS][DI];
  __shared__ __attribute__((aligned(16))) __bf16 x2s[MROWS][DI];
  __shared__ float cbuf[MROWS][HL];
  __shared__ float xes[MROWS];

  const int tid  = threadIdx.x;
  const int lane = tid & 31;
  const int w    = tid >> 5;
  const int bm0  = blockIdx.x * MROWS;
  const int col  = lane & 15;
  const int rb   = (lane >> 4) << 3;   // 0 or 8 (C/D row base)

  // ---- init state: h = c = trg = 0, padding columns = 0 ----
  for (int i = tid; i < MROWS * K1; i += 256) ((__bf16*)cat1)[i] = (__bf16)0.f;
  for (int i = tid; i < MROWS * K2; i += 256) ((__bf16*)cat2)[i] = (__bf16)0.f;
  for (int i = tid; i < MROWS * HL; i += 256) ((float*)cbuf)[i] = 0.f;
  __syncthreads();

  for (int t = 0; t < T_SZ; ++t) {
    // ---------------- phase 0: stage pe(t), emb(t), xe(t); h -> cat1 -------
    for (int i = tid; i < MROWS * HL; i += 256) {
      int m = i / HL, k = i - m * HL;
      cat1[m][H1OFF + k] = cat2[m][k];               // h(t-1) into gate input
    }
    for (int i = tid; i < MROWS * DMD; i += 256) {
      int m = i / DMD, k = i - m * DMD;
      cat1[m][PE0 + k] = (__bf16)pos_table[t * DMD + k];
    }
    for (int i = tid; i < MROWS * (DMD - 1); i += 256) {
      int m = i / (DMD - 1), k = i - m * (DMD - 1);
      cat2[m][EMB0 + k] = (__bf16)emb_seq[((size_t)(bm0 + m) * T_SZ + t) * DMD + k];
    }
    if (tid < MROWS) {
      float v = emb_seq[((size_t)(bm0 + tid) * T_SZ + t) * DMD + (DMD - 1)];
      xes[tid] = __builtin_isinf(v) ? 0.f : v;
    }
    __syncthreads();

    // ---------------- phase 1: gates GEMM (N=2112,K=1120) + LSTM update ----
    for (int jt = w; jt < HL / 16; jt += NW) {       // 33 n-tiles, 4 gates each
      const int n0 = jt * 16;
      v8f ai = {}, af = {}, ag = {}, ao = {};
      for (int kt = 0; kt < K1 / 32; ++kt) {
        const int k0 = kt * 32;
        v16bf a  = wload_a(&cat1[0][0], K1, k0, lane);
        v16bf bi = wload_b(Wg, K1, 0 * HL + n0, k0, lane);
        v16bf bf = wload_b(Wg, K1, 1 * HL + n0, k0, lane);
        v16bf bg = wload_b(Wg, K1, 2 * HL + n0, k0, lane);
        v16bf bo = wload_b(Wg, K1, 3 * HL + n0, k0, lane);
        ai = wmma_bf16(a, bi, ai);
        af = wmma_bf16(a, bf, af);
        ag = wmma_bf16(a, bg, ag);
        ao = wmma_bf16(a, bo, ao);
      }
#pragma unroll
      for (int jj = 0; jj < 8; ++jj) {
        const int m = rb + jj;
        const int n = n0 + col;
        float iv = sigm (ai[jj] + biasg[n]);
        float fv = sigm (af[jj] + biasg[HL + n]);
        float gv = tanhf(ag[jj] + biasg[2 * HL + n]);
        float ov = sigm (ao[jj] + biasg[3 * HL + n]);
        float cn = fv * cbuf[m][n] + iv * gv;
        cbuf[m][n] = cn;
        cat2[m][n] = (__bf16)(ov * tanhf(cn));       // h(t) (cat1 copy next step)
      }
    }
    __syncthreads();

    // ---------------- phase 2: x1 = lrelu([h,emb] @ W1p.T + b1) ------------
    for (int jt = w; jt < DI / 16; jt += NW) {       // 64 n-tiles
      const int n0 = jt * 16;
      v8f acc = {};
      for (int kt = 0; kt < K2 / 32; ++kt) {
        v16bf a = wload_a(&cat2[0][0], K2, kt * 32, lane);
        v16bf b = wload_b(W1p, K2, n0, kt * 32, lane);
        acc = wmma_bf16(a, b, acc);
      }
#pragma unroll
      for (int jj = 0; jj < 8; ++jj) {
        const int m = rb + jj;
        const int n = n0 + col;
        x1s[m][n] = (__bf16)lrelu(acc[jj] + b1[n]);
      }
    }
    __syncthreads();

    // ---------------- phase 3: x2 = x1 + lrelu(x1 @ Wm.T + bm) -------------
    for (int jt = w; jt < DI / 16; jt += NW) {
      const int n0 = jt * 16;
      v8f acc = {};
      for (int kt = 0; kt < DI / 32; ++kt) {
        v16bf a = wload_a(&x1s[0][0], DI, kt * 32, lane);
        v16bf b = wload_b(Wmp, DI, n0, kt * 32, lane);
        acc = wmma_bf16(a, b, acc);
      }
#pragma unroll
      for (int jj = 0; jj < 8; ++jj) {
        const int m = rb + jj;
        const int n = n0 + col;
        float v = lrelu(acc[jj] + bmv[n]);
        x2s[m][n] = (__bf16)((float)x1s[m][n] + v);
      }
    }
    __syncthreads();

    // ---------------- phase 4: out = x2 @ W2.T + b2; trg update ------------
    for (int jt = w; jt < N2P / 16; jt += NW) {      // 17 n-tiles
      const int n0 = jt * 16;
      v8f acc = {};
      for (int kt = 0; kt < DI / 32; ++kt) {
        v16bf a = wload_a(&x2s[0][0], DI, kt * 32, lane);
        v16bf b = wload_b(W2p, DI, n0, kt * 32, lane);
        acc = wmma_bf16(a, b, acc);
      }
#pragma unroll
      for (int jj = 0; jj < 8; ++jj) {
        const int m = rb + jj;
        const int n = n0 + col;
        if (n < DMD) {
          float o = acc[jj] + b2[n];
          if (n == DMD - 1) o = xes[m];              // trg[:, -1] = xe_t
          out[((size_t)(bm0 + m) * T_SZ + t) * DMD + n] = o;
          cat1[m][n] = (__bf16)o;                    // new trg for step t+1
        }
      }
    }
    __syncthreads();
  }
}

// ---------------------------------------------------------------------------
// Launch
// ---------------------------------------------------------------------------
extern "C" void kernel_launch(void* const* d_in, const int* in_sizes, int n_in,
                              void* d_out, int out_size, void* d_ws, size_t ws_size,
                              hipStream_t stream) {
  const float* emb_seq   = (const float*)d_in[0];
  // d_in[1] frame, d_in[2] slf_attn_mask, d_in[3] if_training: unused
  const float* pos_table = (const float*)d_in[4];
  const float* W_ih      = (const float*)d_in[5];
  const float* W_hh      = (const float*)d_in[6];
  const float* b_ih      = (const float*)d_in[7];
  const float* b_hh      = (const float*)d_in[8];
  const float* W1        = (const float*)d_in[9];
  const float* b1        = (const float*)d_in[10];
  const float* Wm        = (const float*)d_in[11];
  const float* bm        = (const float*)d_in[12];
  const float* W2        = (const float*)d_in[13];
  const float* b2        = (const float*)d_in[14];
  float* out = (float*)d_out;

  char* ws = (char*)d_ws;
  size_t off = 0;
  __bf16* Wg  = (__bf16*)(ws + off); off += (size_t)NGATE * K1 * 2;  // 4.73 MB
  __bf16* W1p = (__bf16*)(ws + off); off += (size_t)DI * K2 * 2;     // 1.70 MB
  __bf16* Wmp = (__bf16*)(ws + off); off += (size_t)DI * DI * 2;     // 2.10 MB
  __bf16* W2p = (__bf16*)(ws + off); off += (size_t)N2P * DI * 2;    // 0.56 MB
  float*  bg  = (float*)(ws + off);  off += (size_t)NGATE * 4;       // ~9.1 MB total

  // weight packing (cheap, runs every call; deterministic)
  pack_gates_k<<<(NGATE * K1 + 255) / 256, 256, 0, stream>>>(W_ih, W_hh, Wg);
  pack_w1_k   <<<(DI * K2 + 255) / 256,    256, 0, stream>>>(W1, W1p);
  pack_wm_k   <<<(DI * DI + 255) / 256,    256, 0, stream>>>(Wm, Wmp);
  pack_w2_k   <<<(N2P * DI + 255) / 256,   256, 0, stream>>>(W2, W2p);
  bias_sum_k  <<<(NGATE + 255) / 256,      256, 0, stream>>>(b_ih, b_hh, bg);

  // persistent decoder: 32 WGs x 16 batch rows, full 200-step loop inside
  decode_kernel<<<B_SZ / MROWS, 256, 0, stream>>>(
      emb_seq, pos_table, Wg, W1p, Wmp, W2p, bg, b1, bm, b2, out);
}
